// EquivariantBlock_4140348473949
// MI455X (gfx1250) — matrix operand order repaired and checked by reading, not compiled, not run on hardware
//
#include <hip/hip_runtime.h>
#include <hip/hip_bf16.h>
#include <math.h>

#define N_NODES 20000
#define N_EDGES 640000
#define HD      128
#define K1PAD   288   // 2H+EF = 258 padded up to multiple of 32
#define KN      256   // node MLP fan-in (H + H)
#define TE      64    // edges per block (4 M-tiles of 16)

typedef __attribute__((ext_vector_type(16))) __bf16 v16bf;
typedef __attribute__((ext_vector_type(8)))  float  v8f;

#define WMMA_BF16(A, B, C) \
  __builtin_amdgcn_wmma_f32_16x16x32_bf16(false, (A), false, (B), (short)0, (C), false, false)

__device__ __forceinline__ float silu_f(float x)    { return x / (1.f + __expf(-x)); }
__device__ __forceinline__ float sigmoid_f(float x) { return 1.f / (1.f + __expf(-x)); }

// ---------------------------------------------------------------------------
// Fragment loaders (wave32). Tiles stored row-major bf16, leading dim `ld`
// (even). k0 is a multiple of 32, so both 16-byte chunks are aligned ->
// force b128 loads via uint4.
// A 16x32 bf16 (ISA layout): dword chunks {kb..kb+3} and {kb+8..kb+11}.
// ---------------------------------------------------------------------------
__device__ __forceinline__ v16bf ld_fragA(const __bf16* p, int ld, int k0, unsigned lane) {
  unsigned half = (lane >> 4) & 1u;
  unsigned m    = lane & 15u;
  const unsigned* rowp = (const unsigned*)(p + (size_t)m * (size_t)ld);
  unsigned kb = (unsigned)(k0 + 8 * (int)half) >> 1;   // dword index, multiple of 4
  union { v16bf v; uint4 q[2]; } f;
  f.q[0] = *(const uint4*)(rowp + kb);
  f.q[1] = *(const uint4*)(rowp + kb + 8);
  return f.v;
}

// B 32x16 bf16, stored TRANSPOSED as [n][k] (row = output column):
// same K order as A: dword chunks {kb..kb+3} and {kb+8..kb+11}, kb = (k0+8*half)/2.
__device__ __forceinline__ v16bf ld_fragB(const __bf16* p, int ld, int k0, unsigned lane) {
  unsigned half = (lane >> 4) & 1u;
  unsigned n    = lane & 15u;
  const unsigned* rowp = (const unsigned*)(p + (size_t)n * (size_t)ld);
  unsigned kb = (unsigned)(k0 + 8 * (int)half) >> 1;   // dword index, multiple of 4
  union { v16bf v; uint4 q[2]; } f;
  f.q[0] = *(const uint4*)(rowp + kb);
  f.q[1] = *(const uint4*)(rowp + kb + 8);
  return f.v;
}

// ---------------------------------------------------------------------------
// Weight convert: f32 row-major [K][128] -> bf16 transposed [128][Kpad], zero pad
// ---------------------------------------------------------------------------
__global__ void convert_wT_kernel(const float* __restrict__ W, __bf16* __restrict__ out,
                                  int K, int Kpad) {
  int idx = blockIdx.x * 256 + threadIdx.x;
  if (idx >= HD * Kpad) return;
  int col = idx / Kpad, k = idx - col * Kpad;
  out[idx] = (k < K) ? (__bf16)W[(size_t)k * HD + col] : (__bf16)0.f;
}

// ---------------------------------------------------------------------------
// Edge geometry precompute: radial, coord_diff
// ---------------------------------------------------------------------------
__global__ void precompute_edges_kernel(const float* __restrict__ x,
                                        const int* __restrict__ erow,
                                        const int* __restrict__ ecol,
                                        float* __restrict__ radial,
                                        float* __restrict__ cdiff) {
  int e = blockIdx.x * 256 + threadIdx.x;
  if (e >= N_EDGES) return;
  int r = erow[e], c = ecol[e];
  float dx = x[3 * r + 0] - x[3 * c + 0];
  float dy = x[3 * r + 1] - x[3 * c + 1];
  float dz = x[3 * r + 2] - x[3 * c + 2];
  float rad = dx * dx + dy * dy + dz * dz;
  radial[e] = rad;
  float inv = 1.f / (sqrtf(rad + 1e-8f) + 1.f);   // NORM_CONSTANT = 1
  cdiff[3 * (size_t)e + 0] = dx * inv;
  cdiff[3 * (size_t)e + 1] = dy * inv;
  cdiff[3 * (size_t)e + 2] = dz * inv;
}

// ---------------------------------------------------------------------------
// GCL edge kernel: 64 edges per block, 8 waves each owning a 16-col tile and
// keeping all B fragments (We1: 9, We2: 4) resident in VGPRs across 4 M-tiles.
//   mij = silu(silu(ein@We1+be1)@We2+be2); att = sigmoid(mij@Wa+ba)
//   agg[row] += mij*att*emask   (un-normalized; consumer multiplies by 1/100)
// ---------------------------------------------------------------------------
__global__ __launch_bounds__(256)
void edge_gcl_kernel(const float* __restrict__ h,
                     const int* __restrict__ erow, const int* __restrict__ ecol,
                     const float* __restrict__ radial, const float* __restrict__ eattr,
                     const float* __restrict__ emask,
                     const __bf16* __restrict__ We1T, const float* __restrict__ be1,
                     const __bf16* __restrict__ We2T, const float* __restrict__ be2,
                     const float* __restrict__ Wa, const float* __restrict__ ba,
                     float* __restrict__ agg) {
  __shared__ union {                 // ein is dead after layer-1 -> reuse for mij
    __bf16 ein[TE][K1PAD];           // 36864 B
    float  mij[TE][HD];              // 32768 B
  } u;
  __shared__ __bf16 m1[TE][HD];      // 16384 B
  __shared__ float  attp[TE];
  __shared__ float  attsc[TE];
  __shared__ int    rsh[TE], csh[TE];
  __shared__ float  wash[HD];

  const int tid = threadIdx.x;
  const int e0  = blockIdx.x * TE;

  if (tid < TE) {
    rsh[tid]  = erow[e0 + tid];
    csh[tid]  = ecol[e0 + tid];
    attp[tid] = 0.f;
  }
  if (tid < HD) wash[tid] = Wa[tid];
  __syncthreads();

  for (int idx = tid; idx < TE * K1PAD; idx += 256) {
    int e = idx / K1PAD, c = idx - e * K1PAD;
    float v;
    if (c < HD)             v = h[(size_t)rsh[e] * HD + c];
    else if (c < 2 * HD)    v = h[(size_t)csh[e] * HD + (c - HD)];
    else if (c == 2 * HD)   v = radial[e0 + e];
    else if (c == 2*HD + 1) v = eattr[e0 + e];
    else                    v = 0.f;
    u.ein[e][c] = (__bf16)v;
  }
  __syncthreads();

  const unsigned lane = tid & 31u;
  const int col0 = (tid >> 5) * 16;
  const int col  = col0 + (int)(lane & 15);
  const int half = (int)(lane >> 4);

  // B fragments resident in registers for all 4 M-tiles
  v16bf b1[K1PAD / 32];
#pragma unroll
  for (int ks = 0; ks < K1PAD / 32; ++ks)
    b1[ks] = ld_fragB(We1T + (size_t)col0 * K1PAD, K1PAD, ks * 32, lane);
  v16bf b2[HD / 32];
#pragma unroll
  for (int ks = 0; ks < HD / 32; ++ks)
    b2[ks] = ld_fragB(We2T + (size_t)col0 * HD, HD, ks * 32, lane);
  const float bias1 = be1[col];
  const float bias2 = be2[col];

  // layer 1: 4 M-tiles x 9 K-steps
#pragma unroll
  for (int mt = 0; mt < TE / 16; ++mt) {
    v8f acc = {0.f, 0.f, 0.f, 0.f, 0.f, 0.f, 0.f, 0.f};
#pragma unroll
    for (int ks = 0; ks < K1PAD / 32; ++ks) {
      v16bf a = ld_fragA(&u.ein[mt * 16][0], K1PAD, ks * 32, lane);
      acc = WMMA_BF16(a, b1[ks], acc);
    }
#pragma unroll
    for (int r = 0; r < 8; ++r)
      m1[mt * 16 + r + 8 * half][col] = (__bf16)silu_f(acc[r] + bias1);
  }
  __syncthreads();   // all ein reads complete; u.mij may now overwrite

  // layer 2: 4 M-tiles x 4 K-steps
#pragma unroll
  for (int mt = 0; mt < TE / 16; ++mt) {
    v8f acc = {0.f, 0.f, 0.f, 0.f, 0.f, 0.f, 0.f, 0.f};
#pragma unroll
    for (int ks = 0; ks < HD / 32; ++ks) {
      v16bf a = ld_fragA(&m1[mt * 16][0], HD, ks * 32, lane);
      acc = WMMA_BF16(a, b2[ks], acc);
    }
#pragma unroll
    for (int r = 0; r < 8; ++r)
      u.mij[mt * 16 + r + 8 * half][col] = silu_f(acc[r] + bias2);
  }
  __syncthreads();

  {  // attention logit: 4 threads x 32 cols per edge
    int e = tid >> 2, c0 = (tid & 3) * 32;
    float s = 0.f;
#pragma unroll
    for (int j = 0; j < 32; ++j) s += u.mij[e][c0 + j] * wash[c0 + j];
    atomicAdd(&attp[e], s);
  }
  __syncthreads();
  if (tid < TE) attsc[tid] = sigmoid_f(attp[tid] + ba[0]) * emask[e0 + tid];
  __syncthreads();

  for (int idx = tid; idx < TE * HD; idx += 256) {
    int e = idx >> 7, c = idx & (HD - 1);
    atomicAdd(&agg[(size_t)rsh[e] * HD + c], u.mij[e][c] * attsc[e]);
  }
}

// ---------------------------------------------------------------------------
// Node kernel: h_out = (h + silu(concat(h, agg/100)@Wn1+bn1)@Wn2 + bn2) * mask
// ---------------------------------------------------------------------------
__global__ __launch_bounds__(256)
void node_kernel(const float* __restrict__ h, const float* __restrict__ agg,
                 const float* __restrict__ nmask,
                 const __bf16* __restrict__ Wn1T, const float* __restrict__ bn1,
                 const __bf16* __restrict__ Wn2T, const float* __restrict__ bn2,
                 float* __restrict__ h_out) {
  __shared__ __bf16 nin[16][KN];
  __shared__ __bf16 m1[16][HD];

  const int tid = threadIdx.x;
  const int n0  = blockIdx.x * 16;

  for (int idx = tid; idx < 16 * KN; idx += 256) {
    int nd = idx >> 8, c = idx & (KN - 1);
    float v = (c < HD) ? h[(size_t)(n0 + nd) * HD + c]
                       : agg[(size_t)(n0 + nd) * HD + (c - HD)] * 0.01f;  // /NORM_FACTOR
    nin[nd][c] = (__bf16)v;
  }
  __syncthreads();

  const unsigned lane = tid & 31u;
  const int col0 = (tid >> 5) * 16;
  const int col  = col0 + (int)(lane & 15);
  const int half = (int)(lane >> 4);

  v8f acc = {0.f, 0.f, 0.f, 0.f, 0.f, 0.f, 0.f, 0.f};
#pragma unroll
  for (int ks = 0; ks < KN / 32; ++ks) {
    v16bf a = ld_fragA(&nin[0][0], KN, ks * 32, lane);
    v16bf b = ld_fragB(Wn1T + (size_t)col0 * KN, KN, ks * 32, lane);
    acc = WMMA_BF16(a, b, acc);
  }
  {
    float bias = bn1[col];
#pragma unroll
    for (int r = 0; r < 8; ++r)
      m1[r + 8 * half][col] = (__bf16)silu_f(acc[r] + bias);
  }
  __syncthreads();

  v8f acc2 = {0.f, 0.f, 0.f, 0.f, 0.f, 0.f, 0.f, 0.f};
#pragma unroll
  for (int ks = 0; ks < HD / 32; ++ks) {
    v16bf a = ld_fragA(&m1[0][0], HD, ks * 32, lane);
    v16bf b = ld_fragB(Wn2T + (size_t)col0 * HD, HD, ks * 32, lane);
    acc2 = WMMA_BF16(a, b, acc2);
  }
  {
    float bias = bn2[col];
#pragma unroll
    for (int r = 0; r < 8; ++r) {
      int m = r + 8 * half;
      size_t gi = (size_t)(n0 + m) * HD + col;
      h_out[gi] = (h[gi] + acc2[r] + bias) * nmask[n0 + m];
    }
  }
}

// ---------------------------------------------------------------------------
// Coordinate edge kernel (64 edges/block, register-resident B fragments):
//   scal = silu(silu(ein@W1+b1)@W2+b2)@W3
//   x_agg[row] += coord_diff * scal * emask   (un-normalized)
// ---------------------------------------------------------------------------
__global__ __launch_bounds__(256)
void edge_coord_kernel(const float* __restrict__ h,
                       const int* __restrict__ erow, const int* __restrict__ ecol,
                       const float* __restrict__ radial, const float* __restrict__ eattr,
                       const float* __restrict__ emask, const float* __restrict__ cdiff,
                       const __bf16* __restrict__ W1T, const float* __restrict__ b1v,
                       const __bf16* __restrict__ W2T, const float* __restrict__ b2v,
                       const float* __restrict__ W3,
                       float* __restrict__ xagg) {
  __shared__ union {
    __bf16 ein[TE][K1PAD];
    float  mij[TE][HD];
  } u;
  __shared__ __bf16 m1[TE][HD];
  __shared__ float  scalp[TE];
  __shared__ int    rsh[TE], csh[TE];
  __shared__ float  w3sh[HD];

  const int tid = threadIdx.x;
  const int e0  = blockIdx.x * TE;

  if (tid < TE) {
    rsh[tid]   = erow[e0 + tid];
    csh[tid]   = ecol[e0 + tid];
    scalp[tid] = 0.f;
  }
  if (tid < HD) w3sh[tid] = W3[tid];
  __syncthreads();

  for (int idx = tid; idx < TE * K1PAD; idx += 256) {
    int e = idx / K1PAD, c = idx - e * K1PAD;
    float v;
    if (c < HD)             v = h[(size_t)rsh[e] * HD + c];
    else if (c < 2 * HD)    v = h[(size_t)csh[e] * HD + (c - HD)];
    else if (c == 2 * HD)   v = radial[e0 + e];
    else if (c == 2*HD + 1) v = eattr[e0 + e];
    else                    v = 0.f;
    u.ein[e][c] = (__bf16)v;
  }
  __syncthreads();

  const unsigned lane = tid & 31u;
  const int col0 = (tid >> 5) * 16;
  const int col  = col0 + (int)(lane & 15);
  const int half = (int)(lane >> 4);

  v16bf bw1[K1PAD / 32];
#pragma unroll
  for (int ks = 0; ks < K1PAD / 32; ++ks)
    bw1[ks] = ld_fragB(W1T + (size_t)col0 * K1PAD, K1PAD, ks * 32, lane);
  v16bf bw2[HD / 32];
#pragma unroll
  for (int ks = 0; ks < HD / 32; ++ks)
    bw2[ks] = ld_fragB(W2T + (size_t)col0 * HD, HD, ks * 32, lane);
  const float bias1 = b1v[col];
  const float bias2 = b2v[col];

#pragma unroll
  for (int mt = 0; mt < TE / 16; ++mt) {
    v8f acc = {0.f, 0.f, 0.f, 0.f, 0.f, 0.f, 0.f, 0.f};
#pragma unroll
    for (int ks = 0; ks < K1PAD / 32; ++ks) {
      v16bf a = ld_fragA(&u.ein[mt * 16][0], K1PAD, ks * 32, lane);
      acc = WMMA_BF16(a, bw1[ks], acc);
    }
#pragma unroll
    for (int r = 0; r < 8; ++r)
      m1[mt * 16 + r + 8 * half][col] = (__bf16)silu_f(acc[r] + bias1);
  }
  __syncthreads();

#pragma unroll
  for (int mt = 0; mt < TE / 16; ++mt) {
    v8f acc = {0.f, 0.f, 0.f, 0.f, 0.f, 0.f, 0.f, 0.f};
#pragma unroll
    for (int ks = 0; ks < HD / 32; ++ks) {
      v16bf a = ld_fragA(&m1[mt * 16][0], HD, ks * 32, lane);
      acc = WMMA_BF16(a, bw2[ks], acc);
    }
#pragma unroll
    for (int r = 0; r < 8; ++r)
      u.mij[mt * 16 + r + 8 * half][col] = silu_f(acc[r] + bias2);
  }
  __syncthreads();

  {  // scal dot: 4 threads x 32 cols per edge
    int e = tid >> 2, c0 = (tid & 3) * 32;
    float s = 0.f;
#pragma unroll
    for (int j = 0; j < 32; ++j) s += u.mij[e][c0 + j] * w3sh[c0 + j];
    atomicAdd(&scalp[e], s);
  }
  __syncthreads();

  if (tid < TE * 3) {
    int e = tid / 3, d = tid - 3 * (tid / 3);
    int ge = e0 + e;
    float v = cdiff[3 * (size_t)ge + d] * scalp[e] * emask[ge];
    atomicAdd(&xagg[(size_t)rsh[e] * 3 + d], v);
  }
}

// ---------------------------------------------------------------------------
// Finalize: d_out = [h * mask (N*H) ; (x + xagg/100) * mask (N*3)]
// ---------------------------------------------------------------------------
__global__ void finalize_kernel(const float* __restrict__ h, const float* __restrict__ x,
                                const float* __restrict__ xagg,
                                const float* __restrict__ nmask, float* __restrict__ out) {
  int idx = blockIdx.x * 256 + threadIdx.x;
  const int hn = N_NODES * HD;
  if (idx < hn) {
    out[idx] = h[idx] * nmask[idx >> 7];
  } else if (idx < hn + N_NODES * 3) {
    int j = idx - hn;
    out[idx] = (x[j] + xagg[j] * 0.01f) * nmask[j / 3];
  }
}

// ---------------------------------------------------------------------------
extern "C" void kernel_launch(void* const* d_in, const int* in_sizes, int n_in,
                              void* d_out, int out_size, void* d_ws, size_t ws_size,
                              hipStream_t stream) {
  (void)in_sizes; (void)n_in; (void)out_size; (void)ws_size;

  const float* h0    = (const float*)d_in[0];
  const float* x0    = (const float*)d_in[1];
  const int*   ei    = (const int*)d_in[2];   // [2][E]
  const float* eattr = (const float*)d_in[3];
  const float* nmask = (const float*)d_in[4];
  const float* emask = (const float*)d_in[5];
  // params flattened pytree-sorted: "coord" (W1,W2,W3,b1,b2) then "gcls"[0..1]
  // each gcl sorted: Wa, We1, We2, Wn1, Wn2, ba, be1, be2, bn1, bn2
  const float* cW1 = (const float*)d_in[6];
  const float* cW2 = (const float*)d_in[7];
  const float* cW3 = (const float*)d_in[8];
  const float* cb1 = (const float*)d_in[9];
  const float* cb2 = (const float*)d_in[10];
  const float *gWa[2], *gWe1[2], *gWe2[2], *gWn1[2], *gWn2[2];
  const float *gba[2], *gbe1[2], *gbe2[2], *gbn1[2], *gbn2[2];
  for (int l = 0; l < 2; ++l) {
    int b = 11 + 10 * l;
    gWa[l]  = (const float*)d_in[b + 0];
    gWe1[l] = (const float*)d_in[b + 1];
    gWe2[l] = (const float*)d_in[b + 2];
    gWn1[l] = (const float*)d_in[b + 3];
    gWn2[l] = (const float*)d_in[b + 4];
    gba[l]  = (const float*)d_in[b + 5];
    gbe1[l] = (const float*)d_in[b + 6];
    gbe2[l] = (const float*)d_in[b + 7];
    gbn1[l] = (const float*)d_in[b + 8];
    gbn2[l] = (const float*)d_in[b + 9];
  }

  // workspace carve-out
  char* ws = (char*)d_ws;
  size_t off = 0;
  auto carve = [&](size_t bytes) -> void* {
    void* p = ws + off;
    off = (off + bytes + 255) & ~(size_t)255;
    return p;
  };
  float* hA     = (float*)carve(sizeof(float) * (size_t)N_NODES * HD);
  float* hB     = (float*)carve(sizeof(float) * (size_t)N_NODES * HD);
  float* agg    = (float*)carve(sizeof(float) * (size_t)N_NODES * HD);
  float* radial = (float*)carve(sizeof(float) * (size_t)N_EDGES);
  float* cdiff  = (float*)carve(sizeof(float) * (size_t)N_EDGES * 3);
  float* xagg   = (float*)carve(sizeof(float) * (size_t)N_NODES * 3);
  __bf16 *We1T[2], *We2T[2], *Wn1T[2], *Wn2T[2];
  for (int l = 0; l < 2; ++l) {
    We1T[l] = (__bf16*)carve(2u * K1PAD * HD);
    We2T[l] = (__bf16*)carve(2u * HD * HD);
    Wn1T[l] = (__bf16*)carve(2u * KN * HD);
    Wn2T[l] = (__bf16*)carve(2u * HD * HD);
  }
  __bf16* cW1T = (__bf16*)carve(2u * K1PAD * HD);
  __bf16* cW2T = (__bf16*)carve(2u * HD * HD);

  const int* erow = ei;
  const int* ecol = ei + N_EDGES;

  auto conv = [&](const float* W, __bf16* outp, int K, int Kpad) {
    int tot = HD * Kpad;
    convert_wT_kernel<<<(tot + 255) / 256, 256, 0, stream>>>(W, outp, K, Kpad);
  };
  for (int l = 0; l < 2; ++l) {
    conv(gWe1[l], We1T[l], 2 * HD + 2, K1PAD);
    conv(gWe2[l], We2T[l], HD, HD);
    conv(gWn1[l], Wn1T[l], KN, KN);
    conv(gWn2[l], Wn2T[l], HD, HD);
  }
  conv(cW1, cW1T, 2 * HD + 2, K1PAD);
  conv(cW2, cW2T, HD, HD);

  precompute_edges_kernel<<<(N_EDGES + 255) / 256, 256, 0, stream>>>(x0, erow, ecol, radial, cdiff);

  const int EB = N_EDGES / TE;   // 10000 tiles of 64 edges
  const int NB = N_NODES / 16;   // 1250 tiles

  // GCL layer 0
  hipMemsetAsync(agg, 0, sizeof(float) * (size_t)N_NODES * HD, stream);
  edge_gcl_kernel<<<EB, 256, 0, stream>>>(h0, erow, ecol, radial, eattr, emask,
                                          We1T[0], gbe1[0], We2T[0], gbe2[0],
                                          gWa[0], gba[0], agg);
  node_kernel<<<NB, 256, 0, stream>>>(h0, agg, nmask, Wn1T[0], gbn1[0], Wn2T[0], gbn2[0], hA);

  // GCL layer 1
  hipMemsetAsync(agg, 0, sizeof(float) * (size_t)N_NODES * HD, stream);
  edge_gcl_kernel<<<EB, 256, 0, stream>>>(hA, erow, ecol, radial, eattr, emask,
                                          We1T[1], gbe1[1], We2T[1], gbe2[1],
                                          gWa[1], gba[1], agg);
  node_kernel<<<NB, 256, 0, stream>>>(hA, agg, nmask, Wn1T[1], gbn1[1], Wn2T[1], gbn2[1], hB);

  // Equivariant coordinate update
  hipMemsetAsync(xagg, 0, sizeof(float) * (size_t)N_NODES * 3, stream);
  edge_coord_kernel<<<EB, 256, 0, stream>>>(hB, erow, ecol, radial, eattr, emask, cdiff,
                                            cW1T, cb1, cW2T, cb2, cW3, xagg);

  int total = N_NODES * HD + N_NODES * 3;
  finalize_kernel<<<(total + 255) / 256, 256, 0, stream>>>(hB, x0, xagg, nmask, (float*)d_out);
}
